// MeshCorrSeg_4097398800411
// MI455X (gfx1250) — compile-verified, matrix operand order checked by reading, not run on hardware
//
#include <hip/hip_runtime.h>
#include <math.h>

typedef __attribute__((ext_vector_type(16))) __bf16 v16bf;
typedef __attribute__((ext_vector_type(8)))  float  v8f;

#define NN 10000
#define NE 60000
#define NHEAD 8
#define GEMM_WAVES 8
#define USE_ASYNC_LDS 1

// ---------------------------------------------------------------------------
// Fragment loaders. ISA layout for 16-bit A/B operands of
// v_wmma_f32_16x16x32_bf16: lane&15 = row (A) / col (B); lane>=16 selects
// K-group +8; vector halves 0..7 = K base..base+7, halves 8..15 = K+16..+23.
// Each fragment is two contiguous 16-byte loads.
// ---------------------------------------------------------------------------
__device__ __forceinline__ v16bf g_frag(const __bf16* base, int row, int Kp, int kb) {
  union { uint4 q[2]; v16bf v; } u;
  const __bf16* p = base + (size_t)row * Kp + kb;
  u.q[0] = *(const uint4*)(p);
  u.q[1] = *(const uint4*)(p + 16);
  return u.v;
}

__device__ __forceinline__ v16bf g_frag_guard(const __bf16* base, int row, int Kp,
                                              int kb, bool valid) {
  union { uint4 q[2]; v16bf v; } u;
  if (valid) {
    const __bf16* p = base + (size_t)row * Kp + kb;
    u.q[0] = *(const uint4*)(p);
    u.q[1] = *(const uint4*)(p + 16);
  } else {
    u.q[0] = make_uint4(0u, 0u, 0u, 0u);
    u.q[1] = make_uint4(0u, 0u, 0u, 0u);
  }
  return u.v;
}

// ---------------------------------------------------------------------------
// Block-tiled bf16 WMMA GEMM: C[M,N] = scale*(A[M,Kp] @ BT[N,Kp]^T) + bias[N].
// Grid: (ceil(tiles_n/8), tiles_m); 8 waves/block; each wave owns a 16x64 C
// tile. The 16xKp A tile is shared by all 8 waves: staged once into LDS with
// global_load_async_to_lds_b128 (ASYNCcnt) and re-read as ds_load_b128 frags.
// ---------------------------------------------------------------------------
__global__ __launch_bounds__(256) void gemm_wmma_bf16(
    const __bf16* __restrict__ A, const __bf16* __restrict__ BT,
    const float* __restrict__ bias, float* __restrict__ Cf,
    __bf16* __restrict__ Cb, int M, int N, int Kp, int relu, float scale) {
  __shared__ __bf16 As[16 * 256];   // Kp <= 256 in this model

  int tiles_n = (N + 63) >> 6;
  int tm   = blockIdx.y;
  int wav  = threadIdx.x >> 5;
  int lane = threadIdx.x & 31;
  int tn   = blockIdx.x * GEMM_WAVES + wav;

  // ---- stage shared A tile (16 x Kp) into LDS ----
  {
    int chunks = (16 * Kp) >> 3;                    // 16-byte chunks
    for (int t = threadIdx.x; t < chunks; t += 256) {
      int row = t / (Kp >> 3);
      int kc  = (t % (Kp >> 3)) << 3;
      int grow = tm * 16 + row;
      if (grow > M - 1) grow = M - 1;               // clamp (rows independent)
      const __bf16* gsrc = A + (size_t)grow * Kp + kc;
      __bf16* ldst = As + row * Kp + kc;
#if USE_ASYNC_LDS
      asm volatile("global_load_async_to_lds_b128 %0, %1, off"
                   :: "v"((unsigned)(uintptr_t)ldst),
                      "v"((unsigned long long)(uintptr_t)gsrc)
                   : "memory");
#else
      *(uint4*)ldst = *(const uint4*)gsrc;
#endif
    }
#if USE_ASYNC_LDS
#if __has_builtin(__builtin_amdgcn_s_wait_asynccnt)
    __builtin_amdgcn_s_wait_asynccnt(0);
#else
    asm volatile("s_wait_asynccnt 0" ::: "memory");
#endif
#endif
    __syncthreads();
  }

  if (tn < tiles_n) {
    int r    = lane & 15;
    int kgrp = (lane >> 4) ? 8 : 0;
    int nb   = tn * 64 + r;

    v8f acc0 = {}; v8f acc1 = {}; v8f acc2 = {}; v8f acc3 = {};
    if (tn * 64 + 64 <= N) {
      // fast path: whole 16x64 tile in-bounds, no guards in the hot loop
      for (int k0 = 0; k0 < Kp; k0 += 32) {
        v16bf af = g_frag(As, r, Kp, k0 + kgrp);    // ds_load_b128 x2
        v16bf b0 = g_frag(BT, nb +  0, Kp, k0 + kgrp);
        v16bf b1 = g_frag(BT, nb + 16, Kp, k0 + kgrp);
        v16bf b2 = g_frag(BT, nb + 32, Kp, k0 + kgrp);
        v16bf b3 = g_frag(BT, nb + 48, Kp, k0 + kgrp);
        acc0 = __builtin_amdgcn_wmma_f32_16x16x32_bf16(false, af, false, b0, (short)0, acc0, false, false);
        acc1 = __builtin_amdgcn_wmma_f32_16x16x32_bf16(false, af, false, b1, (short)0, acc1, false, false);
        acc2 = __builtin_amdgcn_wmma_f32_16x16x32_bf16(false, af, false, b2, (short)0, acc2, false, false);
        acc3 = __builtin_amdgcn_wmma_f32_16x16x32_bf16(false, af, false, b3, (short)0, acc3, false, false);
      }
    } else {
      for (int k0 = 0; k0 < Kp; k0 += 32) {
        v16bf af = g_frag(As, r, Kp, k0 + kgrp);
        v16bf b0 = g_frag_guard(BT, nb +  0, Kp, k0 + kgrp, (nb +  0) < N);
        v16bf b1 = g_frag_guard(BT, nb + 16, Kp, k0 + kgrp, (nb + 16) < N);
        v16bf b2 = g_frag_guard(BT, nb + 32, Kp, k0 + kgrp, (nb + 32) < N);
        v16bf b3 = g_frag_guard(BT, nb + 48, Kp, k0 + kgrp, (nb + 48) < N);
        acc0 = __builtin_amdgcn_wmma_f32_16x16x32_bf16(false, af, false, b0, (short)0, acc0, false, false);
        acc1 = __builtin_amdgcn_wmma_f32_16x16x32_bf16(false, af, false, b1, (short)0, acc1, false, false);
        acc2 = __builtin_amdgcn_wmma_f32_16x16x32_bf16(false, af, false, b2, (short)0, acc2, false, false);
        acc3 = __builtin_amdgcn_wmma_f32_16x16x32_bf16(false, af, false, b3, (short)0, acc3, false, false);
      }
    }

    // C/D layout: lanes 0-15 -> rows tm*16+v, lanes 16-31 -> rows +8; col = lane&15.
    int row_hi = (lane >> 4) ? 8 : 0;
    v8f accs[4] = {acc0, acc1, acc2, acc3};
#pragma unroll
    for (int s = 0; s < 4; ++s) {
      int n = tn * 64 + s * 16 + r;
      if (n >= N) continue;
      float bv = bias ? bias[n] : 0.0f;
#pragma unroll
      for (int vv = 0; vv < 8; ++vv) {
        int row = tm * 16 + vv + row_hi;
        if (row >= M) continue;
        float val = accs[s][vv] * scale + bv;
        if (relu) val = fmaxf(val, 0.0f);
        size_t idx = (size_t)row * N + n;
        if (Cf) Cf[idx] = val;
        if (Cb) Cb[idx] = (__bf16)val;
      }
    }
  }
}

// f32 [M,K] -> bf16 [M,Kp] zero-padded
__global__ void cvt_pad_kernel(const float* __restrict__ src, __bf16* __restrict__ dst,
                               int M, int K, int Kp) {
  int i = blockIdx.x * blockDim.x + threadIdx.x;
  if (i >= M * Kp) return;
  int m = i / Kp, k = i % Kp;
  dst[i] = (k < K) ? (__bf16)src[(size_t)m * K + k] : (__bf16)0.0f;
}

// f32 W[K,N] -> bf16 WT[N,Kp] (transpose + pad)
__global__ void wtrans_kernel(const float* __restrict__ W, __bf16* __restrict__ WT,
                              int K, int N, int Kp) {
  int i = blockIdx.x * blockDim.x + threadIdx.x;
  if (i >= N * Kp) return;
  int n = i / Kp, k = i % Kp;
  WT[i] = (k < K) ? (__bf16)W[(size_t)k * N + n] : (__bf16)0.0f;
}

// per-node attention logits base: XU[v, h] = sum_k h[v,k] * U[k,h]
__global__ void xu_kernel(const float* __restrict__ h, const float* __restrict__ U,
                          float* __restrict__ XU, int fin) {
  int v = blockIdx.x * blockDim.x + threadIdx.x;
  if (v >= NN) return;
  float acc[NHEAD];
#pragma unroll
  for (int j = 0; j < NHEAD; ++j) acc[j] = 0.0f;
  const float* hr = h + (size_t)v * fin;
  for (int k = 0; k < fin; ++k) {
    float x = hr[k];
    const float* ur = U + (size_t)k * NHEAD;
#pragma unroll
    for (int j = 0; j < NHEAD; ++j) acc[j] += x * ur[j];
  }
#pragma unroll
  for (int j = 0; j < NHEAD; ++j) XU[(size_t)v * NHEAD + j] = acc[j];
}

// One wave per edge (incl. self loops): softmax over heads, weighted gather
// of the precomputed node projection XW[src], atomic scatter into agg[dst].
__global__ void feast_edge_kernel(const int* __restrict__ ei, const float* __restrict__ XW,
                                  const float* __restrict__ XU, const float* __restrict__ cvec,
                                  int fout, float* __restrict__ agg, float* __restrict__ cnt) {
  int gw = (int)((blockIdx.x * blockDim.x + threadIdx.x) >> 5);
  int lane = threadIdx.x & 31;
  int tot = NE + NN;
  if (gw >= tot) return;
  int s, d;
  if (gw < NE) { s = ei[gw]; d = ei[NE + gw]; }
  else         { s = gw - NE; d = s; }

  float lg[NHEAD];
  float mx = -3.0e38f;
#pragma unroll
  for (int h = 0; h < NHEAD; ++h) {
    lg[h] = XU[(size_t)d * NHEAD + h] - XU[(size_t)s * NHEAD + h] + cvec[h];
    mx = fmaxf(mx, lg[h]);
  }
  float sum = 0.0f;
#pragma unroll
  for (int h = 0; h < NHEAD; ++h) { lg[h] = __expf(lg[h] - mx); sum += lg[h]; }
  float inv = 1.0f / sum;

  const float* xw = XW + (size_t)s * NHEAD * fout;
  for (int o = lane; o < fout; o += 32) {
    float msg = 0.0f;
#pragma unroll
    for (int h = 0; h < NHEAD; ++h) msg += lg[h] * xw[(size_t)h * fout + o];
    atomicAdd(agg + (size_t)d * fout + o, msg * inv);
  }
  if (lane == 0) atomicAdd(cnt + d, 1.0f);
}

// out = agg/cnt + bias (+relu); optional f32 and bf16 outputs
__global__ void feast_finalize_kernel(const float* __restrict__ agg, const float* __restrict__ cnt,
                                      const float* __restrict__ bias, float* __restrict__ outF,
                                      __bf16* __restrict__ outB, int fout, int relu) {
  int i = blockIdx.x * blockDim.x + threadIdx.x;
  if (i >= NN * fout) return;
  int v = i / fout, o = i % fout;
  float val = agg[i] / cnt[v] + bias[o];
  if (relu) val = fmaxf(val, 0.0f);
  if (outF) outF[i] = val;
  if (outB) outB[i] = (__bf16)val;
}

// BatchNorm stats: one block per channel, deterministic tree reduction
__global__ void bn_stats_kernel(const float* __restrict__ x, const float* __restrict__ gamma,
                                const float* __restrict__ beta, float* __restrict__ scale,
                                float* __restrict__ shift, int C) {
  int c = blockIdx.x, t = threadIdx.x;
  float s = 0.0f, sq = 0.0f;
  for (int v = t; v < NN; v += 256) {
    float val = x[(size_t)v * C + c];
    s += val; sq += val * val;
  }
  __shared__ float ls[256], lq[256];
  ls[t] = s; lq[t] = sq; __syncthreads();
  for (int o = 128; o > 0; o >>= 1) {
    if (t < o) { ls[t] += ls[t + o]; lq[t] += lq[t + o]; }
    __syncthreads();
  }
  if (t == 0) {
    float mean = ls[0] / (float)NN;
    float var  = lq[0] / (float)NN - mean * mean;
    float inv  = rsqrtf(var + 1e-5f);
    float sc   = gamma[c] * inv;
    scale[c] = sc;
    shift[c] = beta[c] - mean * sc;
  }
}

__global__ void bn_apply_kernel(const float* __restrict__ x, const float* __restrict__ scale,
                                const float* __restrict__ shift, float* __restrict__ outF,
                                __bf16* __restrict__ outB, int C) {
  int i = blockIdx.x * blockDim.x + threadIdx.x;
  if (i >= NN * C) return;
  int c = i % C;
  float v = x[i] * scale[c] + shift[c];
  outF[i] = v;
  outB[i] = (__bf16)v;
}

// ---- Sinkhorn (log domain), online logsumexp, deterministic reductions ----
__global__ void sk_row_kernel(const float* __restrict__ S, const float* __restrict__ vv,
                              float* __restrict__ u, float log_mu) {
  int i = blockIdx.x, t = threadIdx.x;
  const float* row = S + (size_t)i * NN;
  float mx = -3.0e38f, s = 0.0f;
  for (int j = t; j < NN; j += 256) {
    float val = row[j] + vv[j];
    if (val > mx) { s = s * __expf(mx - val) + 1.0f; mx = val; }
    else s += __expf(val - mx);
  }
  __shared__ float sm[256], ss[256];
  sm[t] = mx; ss[t] = s; __syncthreads();
  for (int o = 128; o > 0; o >>= 1) {
    if (t < o) {
      float m2 = sm[t + o], s2 = ss[t + o];
      float M = fmaxf(sm[t], m2);
      ss[t] = ss[t] * __expf(sm[t] - M) + s2 * __expf(m2 - M);
      sm[t] = M;
    }
    __syncthreads();
  }
  if (t == 0) u[i] = log_mu - (sm[0] + __logf(ss[0]));
}

#define SK_PARTS 16
__global__ void sk_col_part_kernel(const float* __restrict__ S, const float* __restrict__ u,
                                   float* __restrict__ pm, float* __restrict__ ps) {
  int j = blockIdx.x * blockDim.x + threadIdx.x;
  int part = blockIdx.y;
  if (j >= NN) return;
  const int rows = NN / SK_PARTS;
  int r0 = part * rows;
  float mx = -3.0e38f, s = 0.0f;
  for (int i = r0; i < r0 + rows; ++i) {
    float val = S[(size_t)i * NN + j] + u[i];
    if (val > mx) { s = s * __expf(mx - val) + 1.0f; mx = val; }
    else s += __expf(val - mx);
  }
  pm[(size_t)part * NN + j] = mx;
  ps[(size_t)part * NN + j] = s;
}

__global__ void sk_col_comb_kernel(const float* __restrict__ pm, const float* __restrict__ ps,
                                   float* __restrict__ vv, float log_nu) {
  int j = blockIdx.x * blockDim.x + threadIdx.x;
  if (j >= NN) return;
  float mx = -3.0e38f, s = 0.0f;
  for (int p = 0; p < SK_PARTS; ++p) {
    float m2 = pm[(size_t)p * NN + j], s2 = ps[(size_t)p * NN + j];
    float M = fmaxf(mx, m2);
    s = s * __expf(mx - M) + s2 * __expf(m2 - M);
    mx = M;
  }
  vv[j] = log_nu - (mx + __logf(s));
}

__global__ void sk_final_kernel(float* __restrict__ S, const float* __restrict__ u,
                                const float* __restrict__ vv) {
  size_t idx = (size_t)blockIdx.x * blockDim.x + threadIdx.x;
  if (idx >= (size_t)NN * NN) return;
  int i = (int)(idx / NN), j = (int)(idx % NN);
  S[idx] += u[i] + vv[j];
}

// ---------------------------------------------------------------------------
// Host orchestration
// ---------------------------------------------------------------------------
static inline void launch_gemm(const __bf16* A, const __bf16* BT, const float* bias,
                               float* Cf, __bf16* Cb, int M, int N, int Kp,
                               int relu, float scale, hipStream_t stream) {
  int tiles_m = (M + 15) / 16;
  int tiles_n = (N + 63) / 64;
  dim3 grid((tiles_n + GEMM_WAVES - 1) / GEMM_WAVES, tiles_m);
  gemm_wmma_bf16<<<grid, 256, 0, stream>>>(A, BT, bias, Cf, Cb, M, N, Kp, relu, scale);
}

extern "C" void kernel_launch(void* const* d_in, const int* in_sizes, int n_in,
                              void* d_out, int out_size, void* d_ws, size_t ws_size,
                              hipStream_t stream) {
  (void)in_sizes; (void)n_in; (void)out_size; (void)ws_size;

  // Input layout (setup_inputs dict insertion order, params flattened
  // enc -> conv -> bn -> cls -> asg, each leaf in declaration order):
  const float* x_in[2]   = {(const float*)d_in[0], (const float*)d_in[2]};
  const int*   ei_in[2]  = {(const int*)d_in[1],  (const int*)d_in[3]};
  const float* enc_w[2]  = {(const float*)d_in[4],  (const float*)d_in[6]};
  const float* enc_b[2]  = {(const float*)d_in[5],  (const float*)d_in[7]};
  const float* conv_W[3] = {(const float*)d_in[8],  (const float*)d_in[12], (const float*)d_in[16]};
  const float* conv_U[3] = {(const float*)d_in[9],  (const float*)d_in[13], (const float*)d_in[17]};
  const float* conv_c[3] = {(const float*)d_in[10], (const float*)d_in[14], (const float*)d_in[18]};
  const float* conv_b[3] = {(const float*)d_in[11], (const float*)d_in[15], (const float*)d_in[19]};
  const float* bn_g[2]   = {(const float*)d_in[20], (const float*)d_in[22]};
  const float* bn_b[2]   = {(const float*)d_in[21], (const float*)d_in[23]};
  const float* cls_w[3]  = {(const float*)d_in[24], (const float*)d_in[26], (const float*)d_in[28]};
  const float* cls_b[3]  = {(const float*)d_in[25], (const float*)d_in[27], (const float*)d_in[29]};
  const float* asg_w[2]  = {(const float*)d_in[30], (const float*)d_in[32]};
  const float* asg_b[2]  = {(const float*)d_in[31], (const float*)d_in[33]};

  float* out = (float*)d_out;                  // [1e8] assignment
  float* cls_out[2] = {out + 100000000ull, out + 100120000ull};

  // ---- workspace bump allocator ----
  char* ws = (char*)d_ws;
  size_t off = 0;
  auto alloc = [&](size_t bytes) -> void* {
    off = (off + 255) & ~(size_t)255;
    void* p = ws + off;
    off += bytes;
    return p;
  };

  __bf16* wt_enc0  = (__bf16*)alloc((size_t)64 * 32 * 2);
  __bf16* wt_enc1  = (__bf16*)alloc((size_t)128 * 64 * 2);
  __bf16* wt_conv[3];
  int conv_fout[3] = {128, 128, 256};
  for (int l = 0; l < 3; ++l)
    wt_conv[l] = (__bf16*)alloc((size_t)(NHEAD * conv_fout[l]) * 128 * 2);
  __bf16* wt_cls0 = (__bf16*)alloc((size_t)256 * 256 * 2);
  __bf16* wt_cls1 = (__bf16*)alloc((size_t)128 * 256 * 2);
  __bf16* wt_cls2 = (__bf16*)alloc((size_t)12 * 128 * 2);
  __bf16* wt_asg0 = (__bf16*)alloc((size_t)256 * 256 * 2);
  __bf16* wt_asg1 = (__bf16*)alloc((size_t)128 * 256 * 2);

  __bf16* fsb = (__bf16*)alloc((size_t)NN * 128 * 2);
  __bf16* ftb = (__bf16*)alloc((size_t)NN * 128 * 2);
  float*  uvec = (float*)alloc((size_t)NN * 4);
  float*  vvec = (float*)alloc((size_t)NN * 4);
  float*  pm   = (float*)alloc((size_t)SK_PARTS * NN * 4);
  float*  ps   = (float*)alloc((size_t)SK_PARTS * NN * 4);

  __bf16* xb    = (__bf16*)alloc((size_t)NN * 32 * 2);
  __bf16* actB0 = (__bf16*)alloc((size_t)NN * 256 * 2);
  __bf16* actB1 = (__bf16*)alloc((size_t)NN * 256 * 2);
  __bf16* actB2 = (__bf16*)alloc((size_t)NN * 256 * 2);
  float*  hF    = (float*)alloc((size_t)NN * 128 * 4);
  float*  yF    = (float*)alloc((size_t)NN * 256 * 4);
  float*  XW    = (float*)alloc((size_t)NN * 2048 * 4);
  float*  XU    = (float*)alloc((size_t)NN * NHEAD * 4);
  float*  agg   = (float*)alloc((size_t)NN * 256 * 4);
  float*  cnt   = (float*)alloc((size_t)NN * 4);
  float*  bnscale = (float*)alloc(128 * 4);
  float*  bnshift = (float*)alloc(128 * 4);

  auto wtrans = [&](const float* W, __bf16* WT, int K, int N, int Kp) {
    int elems = N * Kp;
    wtrans_kernel<<<(elems + 255) / 256, 256, 0, stream>>>(W, WT, K, N, Kp);
  };

  // ---- stage weights (f32 -> transposed padded bf16), once per call ----
  wtrans(enc_w[0], wt_enc0, 3, 64, 32);
  wtrans(enc_w[1], wt_enc1, 64, 128, 64);
  for (int l = 0; l < 3; ++l) wtrans(conv_W[l], wt_conv[l], 128, NHEAD * conv_fout[l], 128);
  wtrans(cls_w[0], wt_cls0, 256, 256, 256);
  wtrans(cls_w[1], wt_cls1, 256, 128, 256);
  wtrans(cls_w[2], wt_cls2, 128, 12, 128);
  wtrans(asg_w[0], wt_asg0, 256, 256, 256);
  wtrans(asg_w[1], wt_asg1, 256, 128, 256);

  // ---- per-graph pipeline ----
  for (int g = 0; g < 2; ++g) {
    const float* x  = x_in[g];
    const int*   ei = ei_in[g];

    // encoder: 3 -> 64 (relu) -> 128
    cvt_pad_kernel<<<(NN * 32 + 255) / 256, 256, 0, stream>>>(x, xb, NN, 3, 32);
    launch_gemm(xb, wt_enc0, enc_b[0], nullptr, actB0, NN, 64, 32, 1, 1.0f, stream);
    launch_gemm(actB0, wt_enc1, enc_b[1], hF, actB1, NN, 128, 64, 0, 1.0f, stream);

    // FeaStConv stack: h (hF f32 / actB1 bf16), fin = 128 throughout
    for (int l = 0; l < 3; ++l) {
      int fout = conv_fout[l];
      // node-level projection XW = h @ W  (the hoisted heavy GEMM)
      launch_gemm(actB1, wt_conv[l], nullptr, XW, nullptr, NN, NHEAD * fout, 128, 0, 1.0f, stream);
      xu_kernel<<<(NN + 255) / 256, 256, 0, stream>>>(hF, conv_U[l], XU, 128);
      hipMemsetAsync(agg, 0, (size_t)NN * fout * 4, stream);
      hipMemsetAsync(cnt, 0, (size_t)NN * 4, stream);
      {
        int waves = NE + NN;
        int blocks = (waves + 7) / 8;  // 8 waves / 256-thread block
        feast_edge_kernel<<<blocks, 256, 0, stream>>>(ei, XW, XU, conv_c[l], fout, agg, cnt);
      }
      if (l < 2) {
        feast_finalize_kernel<<<(NN * fout + 255) / 256, 256, 0, stream>>>(
            agg, cnt, conv_b[l], yF, nullptr, fout, 1);          // mean + bias + relu
        bn_stats_kernel<<<fout, 256, 0, stream>>>(yF, bn_g[l], bn_b[l], bnscale, bnshift, fout);
        bn_apply_kernel<<<(NN * fout + 255) / 256, 256, 0, stream>>>(
            yF, bnscale, bnshift, hF, actB1, fout);              // new h (f32 + bf16)
      } else {
        feast_finalize_kernel<<<(NN * fout + 255) / 256, 256, 0, stream>>>(
            agg, cnt, conv_b[l], nullptr, actB0, fout, 0);       // emb (bf16), no relu
      }
    }

    // cls head: 256 -> 256 (relu) -> 128 (relu) -> 12
    launch_gemm(actB0, wt_cls0, cls_b[0], nullptr, actB1, NN, 256, 256, 1, 1.0f, stream);
    launch_gemm(actB1, wt_cls1, cls_b[1], nullptr, actB2, NN, 128, 256, 1, 1.0f, stream);
    launch_gemm(actB2, wt_cls2, cls_b[2], cls_out[g], nullptr, NN, 12, 128, 0, 1.0f, stream);

    // asg head: 256 -> 256 (relu) -> 128
    launch_gemm(actB0, wt_asg0, asg_b[0], nullptr, actB1, NN, 256, 256, 1, 1.0f, stream);
    launch_gemm(actB1, wt_asg1, asg_b[1], nullptr, (g == 0) ? fsb : ftb, NN, 128, 256, 0, 1.0f, stream);
  }

  // ---- scores = fs @ ft^T / sqrt(256): ftb is already the [N,K] BT operand ----
  launch_gemm(fsb, ftb, nullptr, out, nullptr, NN, NN, 128, 0, 1.0f / 16.0f, stream);

  // ---- log-Sinkhorn, 5 iterations ----
  const float log_mu = -9.210340371976184f;  // -log(10000)
  hipMemsetAsync(uvec, 0, (size_t)NN * 4, stream);
  hipMemsetAsync(vvec, 0, (size_t)NN * 4, stream);
  for (int it = 0; it < 5; ++it) {
    sk_row_kernel<<<NN, 256, 0, stream>>>(out, vvec, uvec, log_mu);
    dim3 cg((NN + 255) / 256, SK_PARTS);
    sk_col_part_kernel<<<cg, 256, 0, stream>>>(out, uvec, pm, ps);
    sk_col_comb_kernel<<<(NN + 255) / 256, 256, 0, stream>>>(pm, ps, vvec, log_mu);
  }
  {
    size_t tot = (size_t)NN * NN;
    int blocks = (int)((tot + 255) / 256);
    sk_final_kernel<<<blocks, 256, 0, stream>>>(out, uvec, vvec);
  }
}
